// InformerForDDPM_69492570849497
// MI455X (gfx1250) — matrix operand order repaired and checked by reading, not compiled, not run on hardware
//
#include <hip/hip_runtime.h>
#include <hip/hip_bf16.h>

// ---------------------------------------------------------------------------
// Informer encoder forward for MI455X (gfx1250, wave32, WMMA).
// All GEMM operands are bf16 in memory (producers write bf16 shadows once);
// the WMMA GEMM stages tiles with CDNA5 async global->LDS copies.
// ---------------------------------------------------------------------------

typedef __bf16 bf16_t;
typedef bf16_t v16bf __attribute__((ext_vector_type(16)));
typedef bf16_t v8bf  __attribute__((ext_vector_type(8)));
typedef float  v8f   __attribute__((ext_vector_type(8)));
typedef unsigned int v4u __attribute__((ext_vector_type(4)));

#define BM 128
#define BN 128
#define BK 32
#define DMODEL 512
#define NHEADS 8
#define DHEAD  64
#define NB     4   // batch

__device__ __forceinline__ unsigned short f2bf(float f) {
  unsigned int u = __float_as_uint(f);
  u += 0x7FFFu + ((u >> 16) & 1u);      // round-to-nearest-even
  return (unsigned short)(u >> 16);
}

// 16-byte async global -> LDS copy (ASYNCcnt), per lane.
__device__ __forceinline__ void async_cp16(unsigned lds_byte_off, const void* g) {
  asm volatile("global_load_async_to_lds_b128 %0, %1, off"
               :: "v"(lds_byte_off), "v"(g) : "memory");
}
__device__ __forceinline__ void wait_async0() {
  asm volatile("s_wait_asynccnt 0x0" ::: "memory");
}

// ---------------------------------------------------------------------------
// Tile staging into one flat LDS block:
//   bytes [0, 16K)  : A double buffer (2 x 128x32 bf16)
//   bytes [16K, 32K): B double buffer (2 x 128x32 bf16, [n][k])
// FULL tiles use async copies; edge tiles use guarded zero-filling copies.
// bt=1: B is [N][K] row-major; bt=0: B is [K][N] (transpose-copied).
// ---------------------------------------------------------------------------
template <bool FULL>
__device__ __forceinline__ void stage_tiles(
    unsigned short* __restrict__ smem, int buf,
    const unsigned short* __restrict__ Ab, const unsigned short* __restrict__ Bb,
    int bt, int M, int N, int K, int lda, int ldb,
    int m0, int n0, int k0, int tid) {
  const unsigned aByte = (unsigned)buf * 8192u;
  const unsigned bByte = 16384u + (unsigned)buf * 8192u;
  unsigned short* sA = smem + (size_t)buf * 4096;
  unsigned short* sB = smem + 8192 + (size_t)buf * 4096;

  // ---- A tile: [m][k], 8 bf16 (16B) per thread per pass ----
#pragma unroll
  for (int p = 0; p < 2; ++p) {
    int e = (p * 256 + tid) * 8;
    int m = e >> 5, k = e & 31;
    const unsigned short* src = &Ab[(long long)(m0 + m) * lda + (k0 + k)];
    if (FULL) {
      async_cp16(aByte + (unsigned)e * 2u, src);
    } else {
      v4u val; val.x = 0; val.y = 0; val.z = 0; val.w = 0;
      if (m0 + m < M) val = *(const v4u*)src;
      *(v4u*)&sA[e] = val;
    }
  }
  // ---- B tile -> [n][k] ----
  if (bt) {
#pragma unroll
    for (int p = 0; p < 2; ++p) {
      int e = (p * 256 + tid) * 8;
      int n = e >> 5, k = e & 31;
      const unsigned short* src = &Bb[(long long)(n0 + n) * ldb + (k0 + k)];
      if (FULL) {
        async_cp16(bByte + (unsigned)e * 2u, src);
      } else {
        v4u val; val.x = 0; val.y = 0; val.z = 0; val.w = 0;
        if (n0 + n < N) val = *(const v4u*)src;
        *(v4u*)&sB[e] = val;
      }
    }
  } else {  // [K][N] source: vector load along n, transpose into [n][k]
#pragma unroll
    for (int p = 0; p < 2; ++p) {
      int idx = p * 256 + tid;      // 0..511
      int k = idx >> 4;             // 0..31
      int n8 = (idx & 15) * 8;      // 0..120
      if (FULL || (n0 + n8 + 7) < N) {
        v4u v = *(const v4u*)&Bb[(long long)(k0 + k) * ldb + (n0 + n8)];
        const unsigned short* pv = (const unsigned short*)&v;
#pragma unroll
        for (int j = 0; j < 8; ++j) sB[(n8 + j) * BK + k] = pv[j];
      } else {
#pragma unroll
        for (int j = 0; j < 8; ++j)
          sB[(n8 + j) * BK + k] =
              (n0 + n8 + j < N) ? Bb[(long long)(k0 + k) * ldb + (n0 + n8 + j)]
                                : (unsigned short)0;
      }
    }
  }
}

// ---------------------------------------------------------------------------
// GEMM:  C[M,N] = act( scale * A[M,K] x B + bias + resid (+ C) )
// A, B bf16; outputs: optional fp32 Cf and/or bf16 shadow Ch.
// Tiles: 128x128x32; 8 waves (4Mx2N), each wave 32x64 = 2x4 WMMA frags.
// Double-buffered LDS with async global->LDS staging. Batched over z.
// ---------------------------------------------------------------------------
__global__ __launch_bounds__(256)
void gemm_bf16_wmma_kernel(const unsigned short* __restrict__ A,
                           const unsigned short* __restrict__ Bm,
                           const float* __restrict__ bias, const float* __restrict__ resid,
                           float* __restrict__ Cf, unsigned short* __restrict__ Ch,
                           int M, int N, int K, int lda, int ldb, int ldc,
                           long long sA_, long long sB_, long long sC_,
                           int bt, int act, int accum, float scale) {
  __shared__ unsigned short smem[16384];  // 32 KB: A[2][4096] ++ B[2][4096]

  const int tid  = threadIdx.x;
  const int lane = tid & 31;
  const int wid  = tid >> 5;
  const int wm   = wid & 3;   // 0..3 -> 32-row slab
  const int wn   = wid >> 2;  // 0..1 -> 64-col slab
  const int l15  = lane & 15;
  const int kh   = lane >> 4;

  const int z = blockIdx.z;
  const unsigned short* Ab = A + (long long)z * sA_;
  const unsigned short* Bb = Bm + (long long)z * sB_;
  float*          Cfb = Cf ? Cf + (long long)z * sC_ : nullptr;
  unsigned short* Chb = Ch ? Ch + (long long)z * sC_ : nullptr;
  const float*    Rb  = resid ? (resid + (long long)z * sC_) : nullptr;

  const int m0 = blockIdx.y * BM;
  const int n0 = blockIdx.x * BN;
  const bool full = (m0 + BM <= M) && (n0 + BN <= N);

  v8f acc[2][4];
#pragma unroll
  for (int i = 0; i < 2; ++i)
#pragma unroll
    for (int j = 0; j < 4; ++j)
#pragma unroll
      for (int r = 0; r < 8; ++r) acc[i][j][r] = 0.f;

  // prologue stage
  if (full) stage_tiles<true >(smem, 0, Ab, Bb, bt, M, N, K, lda, ldb, m0, n0, 0, tid);
  else      stage_tiles<false>(smem, 0, Ab, Bb, bt, M, N, K, lda, ldb, m0, n0, 0, tid);

  int cur = 0;
  for (int k0 = 0; k0 < K; k0 += BK, cur ^= 1) {
    wait_async0();        // our async copies into 'cur' are done
    __syncthreads();      // ...and visible to all waves
    if (k0 + BK < K) {    // stage next tile into the other buffer
      if (full) stage_tiles<true >(smem, cur ^ 1, Ab, Bb, bt, M, N, K, lda, ldb, m0, n0, k0 + BK, tid);
      else      stage_tiles<false>(smem, cur ^ 1, Ab, Bb, bt, M, N, K, lda, ldb, m0, n0, k0 + BK, tid);
    }
    const unsigned short* sA = smem + (size_t)cur * 4096;
    const unsigned short* sB = smem + 8192 + (size_t)cur * 4096;

    v16bf af[2];
#pragma unroll
    for (int tm = 0; tm < 2; ++tm) {
      // A fragment (16x32 bf16): lanes 0-15 row M=l15 hold K {0..7,16..23},
      // lanes 16-31 hold K {8..15,24..31} -> two contiguous 16B LDS chunks.
      const unsigned short* pa = &sA[(wm * 32 + tm * 16 + l15) * BK + kh * 8];
      v8bf alo = *(const v8bf*)pa;
      v8bf ahi = *(const v8bf*)(pa + 16);
      af[tm] = __builtin_shufflevector(alo, ahi, 0, 1, 2, 3, 4, 5, 6, 7,
                                       8, 9, 10, 11, 12, 13, 14, 15);
    }
#pragma unroll
    for (int tn = 0; tn < 4; ++tn) {
      const unsigned short* pb = &sB[(wn * 64 + tn * 16 + l15) * BK + kh * 8];
      v8bf blo = *(const v8bf*)pb;
      v8bf bhi = *(const v8bf*)(pb + 16);
      v16bf bf = __builtin_shufflevector(blo, bhi, 0, 1, 2, 3, 4, 5, 6, 7,
                                         8, 9, 10, 11, 12, 13, 14, 15);
#pragma unroll
      for (int tm = 0; tm < 2; ++tm)
        acc[tm][tn] = __builtin_amdgcn_wmma_f32_16x16x32_bf16(
            false, af[tm], false, bf, (short)0, acc[tm][tn], false, false);
    }
  }

  // ---- epilogue: D layout = 8 VGPRs, lanes 0-15 M=r, lanes 16-31 M=r+8 ----
#pragma unroll
  for (int tm = 0; tm < 2; ++tm) {
#pragma unroll
    for (int tn = 0; tn < 4; ++tn) {
      int col = n0 + wn * 64 + tn * 16 + l15;
#pragma unroll
      for (int r = 0; r < 8; ++r) {
        int row = m0 + wm * 32 + tm * 16 + r + kh * 8;
        if (full || (row < M && col < N)) {
          long long ci = (long long)row * ldc + col;
          float v = acc[tm][tn][r] * scale;
          if (bias)          v += bias[col];
          if (Rb)            v += Rb[ci];
          if (accum && Cfb)  v += Cfb[ci];
          if (act == 1) v = 0.5f * v * (1.f + erff(v * 0.70710678118f)); // exact GELU
          if (Cfb) Cfb[ci] = v;
          if (Chb) Chb[ci] = f2bf(v);
        }
      }
    }
  }
}

// ---------------------------------------------------------------------------
// Support kernels (tiny vs GEMM cost; all L2-resident traffic)
// ---------------------------------------------------------------------------

// W(O,I,3) -> Wt[t][o][i] (bf16)
__global__ void repack_w_bf16_kernel(const float* __restrict__ W, unsigned short* __restrict__ Wt,
                                     int O, int I) {
  int idx = blockIdx.x * blockDim.x + threadIdx.x;
  int total = 3 * O * I;
  if (idx >= total) return;
  int t = idx / (O * I);
  int rem = idx % (O * I);
  int o = rem / I, ci = rem % I;
  Wt[idx] = f2bf(W[((long long)o * I + ci) * 3 + t]);
}

__global__ void conv_bf16_kernel(const float* __restrict__ src, unsigned short* __restrict__ dst,
                                 int n) {
  int idx = blockIdx.x * blockDim.x + threadIdx.x;
  if (idx < n) dst[idx] = f2bf(src[idx]);
}

// circular pad in length: X(B,Lc,C) fp32 -> Xp(B,Lc+2,C) bf16
__global__ void pad_circ_bf16_kernel(const float* __restrict__ X, unsigned short* __restrict__ Xp,
                                     int Lc, int C) {
  int idx = blockIdx.x * blockDim.x + threadIdx.x;
  int total = NB * (Lc + 2) * C;
  if (idx >= total) return;
  int c = idx % C;
  int l = (idx / C) % (Lc + 2);
  int b = idx / (C * (Lc + 2));
  int sl = (l == 0) ? (Lc - 1) : ((l == Lc + 1) ? 0 : (l - 1));
  Xp[idx] = f2bf(X[((long long)(b * Lc + sl)) * C + c]);
}

__global__ void add_pe_kernel(float* __restrict__ X, unsigned short* __restrict__ Xh, int Lc) {
  int idx = blockIdx.x * blockDim.x + threadIdx.x;
  int total = NB * Lc * DMODEL;
  if (idx >= total) return;
  int c = idx & (DMODEL - 1);
  int l = (idx >> 9) % Lc;
  int j = c >> 1;
  float div = __expf(-(float)(2 * j) * (9.210340371976184f / (float)DMODEL));
  float ang = (float)l * div;
  float v = X[idx] + ((c & 1) ? __cosf(ang) : __sinf(ang));
  X[idx] = v;
  Xh[idx] = f2bf(v);
}

// row LayerNorm over width 512; writes fp32 + bf16 shadow
__global__ __launch_bounds__(256)
void layernorm_kernel(const float* __restrict__ X, const float* __restrict__ g,
                      const float* __restrict__ bb, float* __restrict__ Y,
                      unsigned short* __restrict__ Yh) {
  long long row = blockIdx.x;
  const float* x = X + row * DMODEL;
  float* y = Y + row * DMODEL;
  unsigned short* yh = Yh + row * DMODEL;
  int tid = threadIdx.x;
  float a = x[tid], b2 = x[tid + 256];
  __shared__ float red[256];
  red[tid] = a + b2;
  __syncthreads();
  for (int s = 128; s > 0; s >>= 1) { if (tid < s) red[tid] += red[tid + s]; __syncthreads(); }
  float mu = red[0] * (1.f / DMODEL);
  __syncthreads();
  float da = a - mu, db = b2 - mu;
  red[tid] = da * da + db * db;
  __syncthreads();
  for (int s = 128; s > 0; s >>= 1) { if (tid < s) red[tid] += red[tid + s]; __syncthreads(); }
  float inv = rsqrtf(red[0] * (1.f / DMODEL) + 1e-5f);
  float o0 = da * inv * g[tid] + bb[tid];
  float o1 = db * inv * g[tid + 256] + bb[tid + 256];
  y[tid] = o0; y[tid + 256] = o1;
  yh[tid] = f2bf(o0); yh[tid + 256] = f2bf(o1);
}

// prob-sparse sampling score M[b,h,l]
__global__ void sample_m_kernel(const float* __restrict__ Q, const float* __restrict__ Kc,
                                float* __restrict__ Mval, int Lc, int sk) {
  int idx = blockIdx.x * blockDim.x + threadIdx.x;
  int total = NB * NHEADS * Lc;
  if (idx >= total) return;
  int l = idx % Lc;
  int bh = idx / Lc;
  int h = bh % NHEADS, b = bh / NHEADS;
  const float* qr = Q + ((long long)(b * Lc + l)) * DMODEL + h * DHEAD;
  float mx = -__builtin_inff(), sm = 0.f;
  for (int s = 0; s < sk; ++s) {
    unsigned int r = (unsigned)l * 2654435761u ^ ((unsigned)s * 40503u + 0x9E3779B9u);
    r ^= r >> 16; r *= 2246822519u; r ^= r >> 13;
    int ki = (int)(r % (unsigned)Lc);
    const float* kr = Kc + ((long long)(b * Lc + ki)) * DMODEL + h * DHEAD;
    float d = 0.f;
    for (int t = 0; t < DHEAD; ++t) d += qr[t] * kr[t];
    mx = fmaxf(mx, d);
    sm += d;
  }
  Mval[idx] = mx - sm / (float)Lc;
}

// deterministic top-u selection + bf16 gather of reduced Q rows
__global__ __launch_bounds__(256)
void topk_gather_kernel(float* __restrict__ Mval, const float* __restrict__ Q,
                        int* __restrict__ Mtop, unsigned short* __restrict__ Qred,
                        int Lc, int u) {
  int bh = blockIdx.x;
  int h = bh % NHEADS, b = bh / NHEADS;
  float* mv = Mval + (long long)bh * Lc;
  __shared__ float sv[256];
  __shared__ int   si[256];
  int tid = threadIdx.x;
  for (int t = 0; t < u; ++t) {
    float best = -__builtin_inff();
    int bi = 0x7fffffff;
    for (int l = tid; l < Lc; l += 256) {
      float v = mv[l];
      if (v > best) { best = v; bi = l; }
    }
    sv[tid] = best; si[tid] = bi;
    __syncthreads();
    for (int s = 128; s > 0; s >>= 1) {
      if (tid < s) {
        if (sv[tid + s] > sv[tid] ||
            (sv[tid + s] == sv[tid] && si[tid + s] < si[tid])) {
          sv[tid] = sv[tid + s]; si[tid] = si[tid + s];
        }
      }
      __syncthreads();
    }
    int sel = si[0];
    if (tid == 0) { Mtop[bh * u + t] = sel; mv[sel] = -__builtin_inff(); }
    if (tid < DHEAD)
      Qred[((long long)bh * u + t) * DHEAD + tid] =
          f2bf(Q[((long long)(b * Lc + sel)) * DMODEL + h * DHEAD + tid]);
    __syncthreads();
  }
}

// softmax over rows of length len; fp32 in-place + bf16 shadow
__global__ __launch_bounds__(256)
void softmax_kernel(float* __restrict__ S, unsigned short* __restrict__ Sh, int len) {
  float* row = S + (long long)blockIdx.x * len;
  unsigned short* rowh = Sh + (long long)blockIdx.x * len;
  __shared__ float red[256];
  int tid = threadIdx.x;
  float mx = -__builtin_inff();
  for (int i = tid; i < len; i += 256) mx = fmaxf(mx, row[i]);
  red[tid] = mx;
  __syncthreads();
  for (int s = 128; s > 0; s >>= 1) { if (tid < s) red[tid] = fmaxf(red[tid], red[tid + s]); __syncthreads(); }
  mx = red[0];
  __syncthreads();
  float sm = 0.f;
  for (int i = tid; i < len; i += 256) { float e = __expf(row[i] - mx); row[i] = e; sm += e; }
  red[tid] = sm;
  __syncthreads();
  for (int s = 128; s > 0; s >>= 1) { if (tid < s) red[tid] += red[tid + s]; __syncthreads(); }
  float inv = 1.f / red[0];
  for (int i = tid; i < len; i += 256) {
    float v = row[i] * inv;
    row[i] = v;
    rowh[i] = f2bf(v);
  }
}

// per-(b, c) mean of V over L -> Vm[b*512+c]
__global__ void vmean_kernel(const float* __restrict__ V, float* __restrict__ Vm, int Lc) {
  int idx = blockIdx.x * blockDim.x + threadIdx.x;
  if (idx >= NB * DMODEL) return;
  int c = idx & (DMODEL - 1), b = idx >> 9;
  const float* p = V + (long long)b * Lc * DMODEL + c;
  float s = 0.f;
  for (int l = 0; l < Lc; ++l) s += p[(long long)l * DMODEL];
  Vm[idx] = s / (float)Lc;
}

__global__ void ctx_bcast_kernel(unsigned short* __restrict__ Ctx16, const float* __restrict__ Vm,
                                 int Lc) {
  int idx = blockIdx.x * blockDim.x + threadIdx.x;
  int total = NB * Lc * DMODEL;
  if (idx >= total) return;
  int c = idx & (DMODEL - 1);
  int b = idx / (DMODEL * Lc);
  Ctx16[idx] = f2bf(Vm[b * DMODEL + c]);
}

__global__ void scatter_kernel(const float* __restrict__ Upd, const int* __restrict__ Mtop,
                               unsigned short* __restrict__ Ctx16, int Lc, int u) {
  int idx = blockIdx.x * blockDim.x + threadIdx.x;
  int total = NB * NHEADS * u * DHEAD;
  if (idx >= total) return;
  int d = idx & (DHEAD - 1);
  int t = (idx >> 6) % u;
  int bh = idx / (DHEAD * u);
  int h = bh % NHEADS, b = bh / NHEADS;
  int l = Mtop[bh * u + t];
  Ctx16[((long long)(b * Lc + l)) * DMODEL + h * DHEAD + d] = f2bf(Upd[idx]);
}

// BatchNorm stats per channel; one block per channel
__global__ __launch_bounds__(256)
void bn_stats_kernel(const float* __restrict__ Y, float* __restrict__ stats, int Mr) {
  int c = blockIdx.x, tid = threadIdx.x;
  float s = 0.f, s2 = 0.f;
  for (int r = tid; r < Mr; r += 256) {
    float v = Y[(long long)r * DMODEL + c];
    s += v; s2 += v * v;
  }
  __shared__ float r1[256], r2[256];
  r1[tid] = s; r2[tid] = s2;
  __syncthreads();
  for (int st = 128; st > 0; st >>= 1) {
    if (tid < st) { r1[tid] += r1[tid + st]; r2[tid] += r2[tid + st]; }
    __syncthreads();
  }
  if (tid == 0) {
    float mu = r1[0] / (float)Mr;
    stats[c] = mu;
    stats[DMODEL + c] = r2[0] / (float)Mr - mu * mu;
  }
}

__global__ void bn_elu_kernel(float* __restrict__ Y, const float* __restrict__ stats,
                              const float* __restrict__ g, const float* __restrict__ bb,
                              int total) {
  int idx = blockIdx.x * blockDim.x + threadIdx.x;
  if (idx >= total) return;
  int c = idx & (DMODEL - 1);
  float v = (Y[idx] - stats[c]) * rsqrtf(stats[DMODEL + c] + 1e-5f) * g[c] + bb[c];
  Y[idx] = v > 0.f ? v : expm1f(v);
}

// maxpool k=3 s=2 pad=1 along L; fp32 + bf16 shadow outputs
__global__ void maxpool_kernel(const float* __restrict__ Y, float* __restrict__ O,
                               unsigned short* __restrict__ Oh, int Lin, int Lout) {
  int idx = blockIdx.x * blockDim.x + threadIdx.x;
  int total = NB * Lout * DMODEL;
  if (idx >= total) return;
  int c = idx & (DMODEL - 1);
  int j = (idx >> 9) % Lout;
  int b = idx / (DMODEL * Lout);
  float m = -__builtin_inff();
  for (int t = -1; t <= 1; ++t) {
    int l = 2 * j + t;
    if (l >= 0 && l < Lin) m = fmaxf(m, Y[((long long)(b * Lin + l)) * DMODEL + c]);
  }
  long long oi = ((long long)(b * Lout + j)) * DMODEL + c;
  O[oi] = m;
  Oh[oi] = f2bf(m);
}

// ---------------------------------------------------------------------------
// Host orchestration
// ---------------------------------------------------------------------------
static inline void gemm(hipStream_t s, const unsigned short* A, const unsigned short* Bm,
                        const float* bias, const float* resid,
                        float* Cf, unsigned short* Ch, int M, int N, int K,
                        int lda, int ldb, int ldc,
                        long long sA, long long sB, long long sC, int batch,
                        int bt, int act, int accum, float scale) {
  dim3 g((N + BN - 1) / BN, (M + BM - 1) / BM, batch);
  gemm_bf16_wmma_kernel<<<g, dim3(256), 0, s>>>(A, Bm, bias, resid, Cf, Ch, M, N, K,
                                                lda, ldb, ldc, sA, sB, sC,
                                                bt, act, accum, scale);
}

extern "C" void kernel_launch(void* const* d_in, const int* in_sizes, int n_in,
                              void* d_out, int out_size, void* d_ws, size_t ws_size,
                              hipStream_t stream) {
  const float* x_enc   = (const float*)d_in[0];
  const float* tok_w   = (const float*)d_in[1];
  const float* Wq      = (const float*)d_in[2];
  const float* bq      = (const float*)d_in[3];
  const float* Wk      = (const float*)d_in[4];
  const float* bk      = (const float*)d_in[5];
  const float* Wv      = (const float*)d_in[6];
  const float* bv      = (const float*)d_in[7];
  const float* Wo      = (const float*)d_in[8];
  const float* bo      = (const float*)d_in[9];
  const float* W1      = (const float*)d_in[10];
  const float* b1      = (const float*)d_in[11];
  const float* W2      = (const float*)d_in[12];
  const float* b2      = (const float*)d_in[13];
  const float* ln1_g   = (const float*)d_in[14];
  const float* ln1_b   = (const float*)d_in[15];
  const float* ln2_g   = (const float*)d_in[16];
  const float* ln2_b   = (const float*)d_in[17];
  const float* conv_w  = (const float*)d_in[18];
  const float* conv_b  = (const float*)d_in[19];
  const float* bn_g    = (const float*)d_in[20];
  const float* bn_b    = (const float*)d_in[21];
  const float* normf_g = (const float*)d_in[22];
  const float* normf_b = (const float*)d_in[23];
  const float* proj_w  = (const float*)d_in[24];
  const float* proj_b  = (const float*)d_in[25];
  (void)in_sizes; (void)n_in; (void)out_size; (void)ws_size;

  const int L0 = 2048, DFF = 2048;

  // workspace layout
  float* ws = (float*)d_ws;
  size_t off = 0;
  auto alloc = [&](size_t n) { float* p = ws + off; off += n; return p; };
  auto alloc_h = [&](size_t n) {  // bf16 (ushort), 16B-aligned
    off = (off + 3) & ~(size_t)3;
    unsigned short* p = (unsigned short*)(ws + off);
    off += (n + 1) / 2;
    return p;
  };
  // fp32 masters
  float* X    = alloc((size_t)NB * L0 * DMODEL);
  float* T1   = alloc((size_t)NB * L0 * DMODEL);
  float* Qb   = alloc((size_t)NB * L0 * DMODEL);
  float* Kb   = alloc((size_t)NB * L0 * DMODEL);
  float* Vb   = alloc((size_t)NB * L0 * DMODEL);
  float* SC   = alloc((size_t)NB * NHEADS * 40 * L0);
  float* MV   = alloc((size_t)NB * NHEADS * L0);
  int*   MT   = (int*)alloc((size_t)NB * NHEADS * 40);
  float* UP   = alloc((size_t)NB * NHEADS * 40 * DHEAD);
  float* VM   = alloc((size_t)NB * DMODEL);
  float* BNS  = alloc((size_t)2 * DMODEL);
  // bf16 shadows (GEMM operands)
  unsigned short* X16   = alloc_h((size_t)NB * L0 * DMODEL);
  unsigned short* K16   = alloc_h((size_t)NB * L0 * DMODEL);
  unsigned short* V16   = alloc_h((size_t)NB * L0 * DMODEL);
  unsigned short* CTX16 = alloc_h((size_t)NB * L0 * DMODEL);
  unsigned short* FF16  = alloc_h((size_t)NB * L0 * DFF);
  unsigned short* SC16  = alloc_h((size_t)NB * NHEADS * 40 * L0);
  unsigned short* QR16  = alloc_h((size_t)NB * NHEADS * 40 * DHEAD);
  unsigned short* XP16  = alloc_h((size_t)NB * (L0 + 2) * DMODEL);
  // bf16 weights
  unsigned short* WQB = alloc_h((size_t)2 * DMODEL * DMODEL);
  unsigned short* WKB = alloc_h((size_t)2 * DMODEL * DMODEL);
  unsigned short* WVB = alloc_h((size_t)2 * DMODEL * DMODEL);
  unsigned short* WOB = alloc_h((size_t)2 * DMODEL * DMODEL);
  unsigned short* W1B = alloc_h((size_t)2 * DFF * DMODEL);
  unsigned short* W2B = alloc_h((size_t)2 * DMODEL * DFF);
  unsigned short* PJB = alloc_h((size_t)32 * DMODEL);
  unsigned short* CWT = alloc_h((size_t)3 * DMODEL * DMODEL);
  unsigned short* TWT = alloc_h((size_t)3 * DMODEL * 32);

  auto nb = [](int n) { return (n + 255) / 256; };

  // ---- weight precision conversion (once per call) ------------------------
  conv_bf16_kernel<<<nb(2 * DMODEL * DMODEL), 256, 0, stream>>>(Wq, WQB, 2 * DMODEL * DMODEL);
  conv_bf16_kernel<<<nb(2 * DMODEL * DMODEL), 256, 0, stream>>>(Wk, WKB, 2 * DMODEL * DMODEL);
  conv_bf16_kernel<<<nb(2 * DMODEL * DMODEL), 256, 0, stream>>>(Wv, WVB, 2 * DMODEL * DMODEL);
  conv_bf16_kernel<<<nb(2 * DMODEL * DMODEL), 256, 0, stream>>>(Wo, WOB, 2 * DMODEL * DMODEL);
  conv_bf16_kernel<<<nb(2 * DFF * DMODEL), 256, 0, stream>>>(W1, W1B, 2 * DFF * DMODEL);
  conv_bf16_kernel<<<nb(2 * DFF * DMODEL), 256, 0, stream>>>(W2, W2B, 2 * DFF * DMODEL);
  conv_bf16_kernel<<<nb(32 * DMODEL), 256, 0, stream>>>(proj_w, PJB, 32 * DMODEL);
  repack_w_bf16_kernel<<<nb(3 * DMODEL * 32), 256, 0, stream>>>(tok_w, TWT, DMODEL, 32);
  repack_w_bf16_kernel<<<nb(3 * DMODEL * DMODEL), 256, 0, stream>>>(conv_w, CWT, DMODEL, DMODEL);

  // ---- token embedding: circular conv (32->512) as 3 tap GEMMs, + PE ------
  pad_circ_bf16_kernel<<<nb(NB * (L0 + 2) * 32), 256, 0, stream>>>(x_enc, XP16, L0, 32);
  for (int t = 0; t < 3; ++t)
    gemm(stream, XP16 + (size_t)t * 32, TWT + (size_t)t * DMODEL * 32, nullptr, nullptr,
         X, nullptr, L0, DMODEL, 32, 32, 32, DMODEL,
         (long long)(L0 + 2) * 32, 0, (long long)L0 * DMODEL, NB,
         /*bt*/1, /*act*/0, /*accum*/t > 0, 1.f);
  add_pe_kernel<<<nb(NB * L0 * DMODEL), 256, 0, stream>>>(X, X16, L0);

  // ---- encoder layers -----------------------------------------------------
  for (int i = 0; i < 2; ++i) {
    const int Lc = i ? 1024 : 2048;
    const int Mr = NB * Lc;
    const int u  = i ? 35 : 40;          // FACTOR * ceil(ln(Lc))
    const int sk = u;
    const unsigned short* WQi = WQB + (size_t)i * DMODEL * DMODEL; const float* bqi = bq + (size_t)i * DMODEL;
    const unsigned short* WKi = WKB + (size_t)i * DMODEL * DMODEL; const float* bki = bk + (size_t)i * DMODEL;
    const unsigned short* WVi = WVB + (size_t)i * DMODEL * DMODEL; const float* bvi = bv + (size_t)i * DMODEL;
    const unsigned short* WOi = WOB + (size_t)i * DMODEL * DMODEL; const float* boi = bo + (size_t)i * DMODEL;
    const unsigned short* W1i = W1B + (size_t)i * DFF * DMODEL;    const float* b1i = b1 + (size_t)i * DFF;
    const unsigned short* W2i = W2B + (size_t)i * DMODEL * DFF;    const float* b2i = b2 + (size_t)i * DMODEL;

    // Q/K/V projections (K,V also produce bf16 shadows for attention GEMMs)
    gemm(stream, X16, WQi, bqi, nullptr, Qb, nullptr, Mr, DMODEL, DMODEL, DMODEL, DMODEL, DMODEL, 0, 0, 0, 1, 1, 0, 0, 1.f);
    gemm(stream, X16, WKi, bki, nullptr, Kb, K16,     Mr, DMODEL, DMODEL, DMODEL, DMODEL, DMODEL, 0, 0, 0, 1, 1, 0, 0, 1.f);
    gemm(stream, X16, WVi, bvi, nullptr, Vb, V16,     Mr, DMODEL, DMODEL, DMODEL, DMODEL, DMODEL, 0, 0, 0, 1, 1, 0, 0, 1.f);

    // prob-sparse query scoring + top-u selection
    sample_m_kernel<<<nb(NB * NHEADS * Lc), 256, 0, stream>>>(Qb, Kb, MV, Lc, sk);
    topk_gather_kernel<<<NB * NHEADS, 256, 0, stream>>>(MV, Qb, MT, QR16, Lc, u);

    // scores = Qred @ K^T / sqrt(d)   (per head, strided bf16 K view)
    for (int bh = 0; bh < NB * NHEADS; ++bh) {
      int b = bh / NHEADS, h = bh % NHEADS;
      gemm(stream, QR16 + (size_t)bh * u * DHEAD,
           K16 + (size_t)b * Lc * DMODEL + (size_t)h * DHEAD, nullptr, nullptr,
           SC + (size_t)bh * u * Lc, nullptr, u, Lc, DHEAD, DHEAD, DMODEL, Lc,
           0, 0, 0, 1, /*bt*/1, 0, 0, 0.125f);
    }
    softmax_kernel<<<NB * NHEADS * u, 256, 0, stream>>>(SC, SC16, Lc);

    // upd = attn @ V   (per head, strided bf16 V view, bt=0)
    for (int bh = 0; bh < NB * NHEADS; ++bh) {
      int b = bh / NHEADS, h = bh % NHEADS;
      gemm(stream, SC16 + (size_t)bh * u * Lc,
           V16 + (size_t)b * Lc * DMODEL + (size_t)h * DHEAD, nullptr, nullptr,
           UP + (size_t)bh * u * DHEAD, nullptr, u, DHEAD, Lc, Lc, DMODEL, DHEAD,
           0, 0, 0, 1, /*bt*/0, 0, 0, 1.f);
    }

    // ctx = broadcast(mean V) (bf16), then scatter updated rows
    vmean_kernel<<<nb(NB * DMODEL), 256, 0, stream>>>(Vb, VM, Lc);
    ctx_bcast_kernel<<<nb(Mr * DMODEL), 256, 0, stream>>>(CTX16, VM, Lc);
    scatter_kernel<<<nb(NB * NHEADS * u * DHEAD), 256, 0, stream>>>(UP, MT, CTX16, Lc, u);

    // O projection + residual, LN1
    gemm(stream, CTX16, WOi, boi, X, T1, nullptr, Mr, DMODEL, DMODEL, DMODEL, DMODEL, DMODEL, 0, 0, 0, 1, 1, 0, 0, 1.f);
    layernorm_kernel<<<Mr, 256, 0, stream>>>(T1, ln1_g + (size_t)i * DMODEL, ln1_b + (size_t)i * DMODEL, X, X16);

    // FFN: GELU(x W1^T + b1) (bf16 only) -> W2^T + b2 + residual, LN2
    gemm(stream, X16, W1i, b1i, nullptr, nullptr, FF16, Mr, DFF, DMODEL, DMODEL, DMODEL, DFF, 0, 0, 0, 1, 1, /*gelu*/1, 0, 1.f);
    gemm(stream, FF16, W2i, b2i, X, T1, nullptr, Mr, DMODEL, DFF, DFF, DFF, DMODEL, 0, 0, 0, 1, 1, 0, 0, 1.f);
    layernorm_kernel<<<Mr, 256, 0, stream>>>(T1, ln2_g + (size_t)i * DMODEL, ln2_b + (size_t)i * DMODEL, X, X16);

    // conv distil after layer 0: circ conv (3 tap GEMMs) + BN + ELU + maxpool
    if (i == 0) {
      pad_circ_bf16_kernel<<<nb(NB * (Lc + 2) * DMODEL), 256, 0, stream>>>(X, XP16, Lc, DMODEL);
      for (int t = 0; t < 3; ++t)
        gemm(stream, XP16 + (size_t)t * DMODEL, CWT + (size_t)t * DMODEL * DMODEL,
             t == 0 ? conv_b : nullptr, nullptr, T1, nullptr, Lc, DMODEL, DMODEL,
             DMODEL, DMODEL, DMODEL,
             (long long)(Lc + 2) * DMODEL, 0, (long long)Lc * DMODEL, NB,
             1, 0, t > 0, 1.f);
      bn_stats_kernel<<<DMODEL, 256, 0, stream>>>(T1, BNS, NB * Lc);
      bn_elu_kernel<<<nb(NB * Lc * DMODEL), 256, 0, stream>>>(T1, BNS, bn_g, bn_b, NB * Lc * DMODEL);
      maxpool_kernel<<<nb(NB * (Lc / 2) * DMODEL), 256, 0, stream>>>(T1, X, X16, Lc, Lc / 2);
    }
  }

  // ---- final LN + projection (writes d_out directly) ----------------------
  const int Lf = 1024, Mf = NB * Lf;
  layernorm_kernel<<<Mf, 256, 0, stream>>>(X, normf_g, normf_b, T1, X16);
  gemm(stream, X16, PJB, proj_b, nullptr, (float*)d_out, nullptr, Mf, 32, DMODEL,
       DMODEL, DMODEL, 32, 0, 0, 0, 1, 1, 0, 0, 1.f);
}
